// MoEAdapter_89945205113232
// MI455X (gfx1250) — matrix operand order repaired and checked by reading, not compiled, not run on hardware
//
#include <hip/hip_runtime.h>
#include <hip/hip_bf16.h>
#include <math.h>

typedef __bf16 bf16;
typedef __attribute__((ext_vector_type(16))) __bf16 v16bf;
typedef __attribute__((ext_vector_type(8)))  __bf16 v8bf;
typedef __attribute__((ext_vector_type(8)))  float  v8f;

#define N_ROWS   8192
#define ID_DIM   128
#define LLM_DIM  4096
#define DIM      4224          // ID_DIM + LLM_DIM
#define OUTD     512
#define HID      1024
#define NEXP     8
#define GHID     16
#define ROWS     64            // rows per block tile
#define HC       256           // h-columns per outer chunk
#define KS       32            // WMMA K step (bf16)

union Frag { v16bf v; v8bf h[2]; };

// ---- CDNA5 helpers -------------------------------------------------------

// LDS byte offset of a generic pointer to __shared__ memory (addrspacecast to AS3).
static __device__ __forceinline__ unsigned ldsAddr(const void* p) {
  return (unsigned)(unsigned long long)(const __attribute__((address_space(3))) void*)p;
}

// Async global->LDS 16B copy (per-lane), tracked by ASYNCcnt.
static __device__ __forceinline__ void asyncLoadB128(unsigned lds_off, const void* gaddr) {
  asm volatile("global_load_async_to_lds_b128 %0, %1, off"
               :: "v"(lds_off), "v"(gaddr) : "memory");
}

static __device__ __forceinline__ void waitAsync0() {
  asm volatile("s_wait_asynccnt 0x0" ::: "memory");
}

static __device__ __forceinline__ v8f wmma_bf16(const Frag& a, const Frag& b, v8f c) {
  return __builtin_amdgcn_wmma_f32_16x16x32_bf16(false, a.v, false, b.v,
                                                 (short)0, c, false, false);
}

// ---- small prep kernels --------------------------------------------------

__global__ __launch_bounds__(256)
void concat_to_bf16(const float* __restrict__ id_emb, const float* __restrict__ llm_emb,
                    bf16* __restrict__ xb) {
  const int d = blockIdx.x * 256 + threadIdx.x;
  const int n = blockIdx.y;
  if (d < DIM) {
    float v = (d < ID_DIM) ? id_emb[(size_t)n * ID_DIM + d]
                           : llm_emb[(size_t)n * LLM_DIM + (d - ID_DIM)];
    xb[(size_t)n * DIM + d] = (bf16)v;
  }
}

// src [z][R][C] f32 -> dst [z][C][R] bf16 ; R,C multiples of 32. block = (32,8)
__global__ __launch_bounds__(256)
void transpose_f32_to_bf16(const float* __restrict__ src, bf16* __restrict__ dst,
                           int R, int C) {
  __shared__ float tile[32][33];
  const int z = blockIdx.z;
  src += (size_t)z * R * C;
  dst += (size_t)z * R * C;
  const int tx = threadIdx.x, ty = threadIdx.y;
  const int c0 = blockIdx.x * 32, r0 = blockIdx.y * 32;
  #pragma unroll
  for (int j = 0; j < 32; j += 8)
    tile[ty + j][tx] = src[(size_t)(r0 + ty + j) * C + c0 + tx];
  __syncthreads();
  #pragma unroll
  for (int j = 0; j < 32; j += 8)
    dst[(size_t)(c0 + ty + j) * R + r0 + tx] = (bf16)tile[tx][ty + j];
}

__global__ __launch_bounds__(256)
void zero_f32(float* __restrict__ p, int count) {
  int i = blockIdx.x * 256 + threadIdx.x;
  if (i < count) p[i] = 0.f;
}

// ---- gating: one wave32 per row -----------------------------------------

__global__ __launch_bounds__(256)
void gate_kernel(const float* __restrict__ id_emb, const float* __restrict__ llm_emb,
                 const float* __restrict__ Wg1, const float* __restrict__ bg1,
                 const float* __restrict__ Wg2, const float* __restrict__ bg2,
                 float* __restrict__ dw) {
  const int lane = threadIdx.x & 31;
  const int wv   = threadIdx.x >> 5;
  const int n    = blockIdx.x * 8 + wv;

  float acc[GHID];
  #pragma unroll
  for (int j = 0; j < GHID; ++j) acc[j] = 0.f;

  for (int d = lane; d < DIM; d += 32) {
    float xv = (d < ID_DIM) ? id_emb[(size_t)n * ID_DIM + d]
                            : llm_emb[(size_t)n * LLM_DIM + (d - ID_DIM)];
    const float* wrow = Wg1 + (size_t)d * GHID;
    #pragma unroll
    for (int j = 0; j < GHID; ++j) acc[j] += xv * wrow[j];
  }
  #pragma unroll
  for (int off = 16; off >= 1; off >>= 1) {
    #pragma unroll
    for (int j = 0; j < GHID; ++j) acc[j] += __shfl_xor(acc[j], off, 32);
  }
  if (lane == 0) {
    float g[GHID];
    #pragma unroll
    for (int j = 0; j < GHID; ++j) g[j] = fmaxf(acc[j] + bg1[j], 0.f);
    float lg[NEXP];
    #pragma unroll
    for (int e = 0; e < NEXP; ++e) {
      float s = bg2[e];
      #pragma unroll
      for (int j = 0; j < GHID; ++j) s += g[j] * Wg2[j * NEXP + e];
      lg[e] = s;
    }
    int i0 = 0; float v0 = lg[0];
    #pragma unroll
    for (int e = 1; e < NEXP; ++e) if (lg[e] > v0) { v0 = lg[e]; i0 = e; }
    int i1 = -1; float v1 = -3.4e38f;
    #pragma unroll
    for (int e = 0; e < NEXP; ++e) if (e != i0 && lg[e] > v1) { v1 = lg[e]; i1 = e; }
    float w0 = 1.f / (1.f + __expf(v1 - v0));   // softmax over top-2 (v0 >= v1)
    float w1 = 1.f - w0;
    #pragma unroll
    for (int e = 0; e < NEXP; ++e) dw[(size_t)n * NEXP + e] = 0.f;
    dw[(size_t)n * NEXP + i0] = w0;
    dw[(size_t)n * NEXP + i1] = w1;
  }
}

// ---- fused expert FFN: relu(x @ W1[e] + b1) @ W2[e] + b2, weighted combine

__global__ __launch_bounds__(256)
void moe_ffn_kernel(const bf16* __restrict__ xb, const bf16* __restrict__ w1t,
                    const bf16* __restrict__ w2t, const float* __restrict__ b1,
                    const float* __restrict__ b2, const float* __restrict__ dw,
                    float* __restrict__ out) {
  // LDS: xs double buffer (8KB) + region shared between W1 staging (2x16KB)
  // in Phase A and the bf16 h-chunk (32KB) in Phase B. Total 40KB.
  __shared__ __align__(16) char smem[2 * ROWS * KS * 2 + 2 * HC * KS * 2];
  bf16* xs  = (bf16*)smem;                              // [2][ROWS*KS]
  bf16* wsA = (bf16*)(smem + 2 * ROWS * KS * 2);        // [2][HC*KS]
  bf16* hs  = (bf16*)(smem + 2 * ROWS * KS * 2);        // [ROWS*HC] (aliases wsA)

  const int tid   = threadIdx.x;
  const int lane  = tid & 31;
  const int w     = tid >> 5;
  const int rsub  = w & 3;            // 4 row sub-tiles of 16
  const int chalf = w >> 2;           // 2 column halves
  const int rbase = rsub * 16;
  const int e     = blockIdx.y;
  const int R0    = blockIdx.x * ROWS;

  const int mcol = lane & 15;
  const int hiA  = (lane < 16) ? 0 : 8;    // unused; A frag also uses hiB
  const int hiB  = (lane < 16) ? 0 : 16;   // B-frag K base offset
  const int mrow = (lane < 16) ? 0 : 8;    // C/D row offset

  const unsigned xsOff[2]  = { ldsAddr(xs),  ldsAddr(xs + ROWS * KS) };
  const unsigned wsOff[2]  = { ldsAddr(wsA), ldsAddr(wsA + HC * KS) };

  const int xr = tid >> 2, xc = tid & 3;   // xs: 64 rows x 4 chunks of 16B

  const v8f vzero = {0.f,0.f,0.f,0.f,0.f,0.f,0.f,0.f};
  v8f oacc[16];
  #pragma unroll
  for (int t = 0; t < 16; ++t) oacc[t] = vzero;

  const size_t w1eBase = (size_t)e * HID * DIM;
  const size_t w2eBase = (size_t)e * OUTD * HID;

  for (int hc = 0; hc < HID / HC; ++hc) {
    const int h0 = hc * HC;
    v8f acc[8];
    #pragma unroll
    for (int t = 0; t < 8; ++t) acc[t] = vzero;

    // prologue: stage K-step 0 into buffer 0
    {
      asyncLoadB128(xsOff[0] + (unsigned)((xr * KS + xc * 8) * 2),
                    xb + (size_t)(R0 + xr) * DIM + xc * 8);
      #pragma unroll
      for (int i = 0; i < 4; ++i) {
        int ch = tid + i * 256; int n = ch >> 2; int cc = ch & 3;
        asyncLoadB128(wsOff[0] + (unsigned)((n * KS + cc * 8) * 2),
                      w1t + w1eBase + (size_t)(h0 + n) * DIM + cc * 8);
      }
    }

    // ---- Phase A: h_chunk[64,256] += x[64,D] @ W1t[e][h0:h0+256, D] ------
    for (int kb = 0; kb < DIM / KS; ++kb) {
      const int buf = kb & 1;
      waitAsync0();
      __syncthreads();
      if (kb + 1 < DIM / KS) {
        const int d0n = (kb + 1) * KS;
        const unsigned xd = xsOff[buf ^ 1];
        const unsigned wd = wsOff[buf ^ 1];
        asyncLoadB128(xd + (unsigned)((xr * KS + xc * 8) * 2),
                      xb + (size_t)(R0 + xr) * DIM + d0n + xc * 8);
        #pragma unroll
        for (int i = 0; i < 4; ++i) {
          int ch = tid + i * 256; int n = ch >> 2; int cc = ch & 3;
          asyncLoadB128(wd + (unsigned)((n * KS + cc * 8) * 2),
                        w1t + w1eBase + (size_t)(h0 + n) * DIM + d0n + cc * 8);
        }
      }
      // A fragment: rows rbase..rbase+15 of x tile
      const bf16* xp = xs + buf * (ROWS * KS) + (rbase + mcol) * KS;
      Frag a; a.h[0] = *(const v8bf*)(xp + hiB); a.h[1] = *(const v8bf*)(xp + hiB + 8);
      #pragma unroll
      for (int t = 0; t < 8; ++t) {
        const bf16* wp = wsA + buf * (HC * KS) + (chalf * 128 + t * 16 + mcol) * KS + hiB;
        Frag b; b.h[0] = *(const v8bf*)wp; b.h[1] = *(const v8bf*)(wp + 8);
        acc[t] = wmma_bf16(a, b, acc[t]);
      }
    }

    // hs aliases wsA -> make sure every wave finished reading wsA first
    __syncthreads();
    #pragma unroll
    for (int t = 0; t < 8; ++t) {
      const int n = chalf * 128 + t * 16 + mcol;
      const float b1v = b1[e * HID + h0 + n];
      #pragma unroll
      for (int r = 0; r < 8; ++r) {
        float hv = acc[t][r] + b1v;
        hs[(rbase + mrow + r) * HC + n] = (bf16)fmaxf(hv, 0.f);
      }
    }
    __syncthreads();

    // ---- Phase B: out[64,512] += h_chunk[64,256] @ W2t[e][:,h0:h0+256] ---
    for (int kk = 0; kk < HC / KS; ++kk) {
      const bf16* hp = hs + (rbase + mcol) * HC + kk * KS;
      Frag a; a.h[0] = *(const v8bf*)(hp + hiB); a.h[1] = *(const v8bf*)(hp + hiB + 8);
      #pragma unroll
      for (int t = 0; t < 16; ++t) {
        const int n = chalf * 256 + t * 16 + mcol;
        const bf16* qp = w2t + w2eBase + (size_t)n * HID + h0 + kk * KS + hiB;
        Frag b; b.h[0] = *(const v8bf*)qp; b.h[1] = *(const v8bf*)(qp + 8);
        oacc[t] = wmma_bf16(a, b, oacc[t]);
      }
    }
    __syncthreads();   // hs reads done before next chunk restages wsA
  }

  // ---- epilogue: out[n,:] += dense_w[n,e] * (expert_out + b2) ------------
  float wvr[8];
  #pragma unroll
  for (int r = 0; r < 8; ++r)
    wvr[r] = dw[(size_t)(R0 + rbase + mrow + r) * NEXP + e];
  #pragma unroll
  for (int t = 0; t < 16; ++t) {
    const int n = chalf * 256 + t * 16 + mcol;
    const float b2v = b2[e * OUTD + n];
    #pragma unroll
    for (int r = 0; r < 8; ++r) {
      atomicAdd(&out[(size_t)(R0 + rbase + mrow + r) * OUTD + n],
                wvr[r] * (oacc[t][r] + b2v));
    }
  }
}

// ---- host side -----------------------------------------------------------

extern "C" void kernel_launch(void* const* d_in, const int* in_sizes, int n_in,
                              void* d_out, int out_size, void* d_ws, size_t ws_size,
                              hipStream_t stream) {
  (void)in_sizes; (void)n_in; (void)out_size; (void)ws_size;
  const float* id_emb = (const float*)d_in[0];
  const float* llm    = (const float*)d_in[1];
  const float* Wg1    = (const float*)d_in[2];
  const float* bg1    = (const float*)d_in[3];
  const float* Wg2    = (const float*)d_in[4];
  const float* bg2    = (const float*)d_in[5];
  const float* W1     = (const float*)d_in[6];
  const float* b1     = (const float*)d_in[7];
  const float* W2     = (const float*)d_in[8];
  const float* b2     = (const float*)d_in[9];
  float* out = (float*)d_out;

  // workspace layout (all 16B aligned): ~147 MB total
  bf16* xb   = (bf16*)d_ws;                          // [N, D] bf16
  bf16* w1t  = xb  + (size_t)N_ROWS * DIM;           // [E, H, D] bf16 (transposed)
  bf16* w2t  = w1t + (size_t)NEXP * HID * DIM;       // [E, OUT, H] bf16 (transposed)
  float* dwp = (float*)(w2t + (size_t)NEXP * OUTD * HID);  // [N, E] f32

  concat_to_bf16<<<dim3((DIM + 255) / 256, N_ROWS), 256, 0, stream>>>(id_emb, llm, xb);
  transpose_f32_to_bf16<<<dim3(HID / 32, DIM / 32, NEXP), dim3(32, 8), 0, stream>>>(
      W1, w1t, DIM, HID);
  transpose_f32_to_bf16<<<dim3(OUTD / 32, HID / 32, NEXP), dim3(32, 8), 0, stream>>>(
      W2, w2t, HID, OUTD);
  gate_kernel<<<N_ROWS / 8, 256, 0, stream>>>(id_emb, llm, Wg1, bg1, Wg2, bg2, dwp);
  zero_f32<<<(N_ROWS * OUTD + 255) / 256, 256, 0, stream>>>(out, N_ROWS * OUTD);
  moe_ffn_kernel<<<dim3(N_ROWS / ROWS, NEXP), 256, 0, stream>>>(
      xb, w1t, w2t, b1, b2, dwp, out);
}
